// Attention_2_687194768263
// MI455X (gfx1250) — compile-verified, matrix operand order checked by reading, not run on hardware
//
#include <hip/hip_runtime.h>
#include <hip/hip_bf16.h>

typedef __attribute__((ext_vector_type(16))) _Float16 v16h;
typedef __attribute__((ext_vector_type(8)))  float    v8f;

#define NB     4
#define HEADS  8
#define DHEAD  32
#define NSEQ   2048
#define DMODEL 256

// Convert 16 contiguous f32 -> v16h (A/B fragment slice for one lane)
__device__ __forceinline__ v16h ld16_f32_to_f16(const float* __restrict__ p) {
    const float4* p4 = (const float4*)p;
    v16h r;
#pragma unroll
    for (int i = 0; i < 4; ++i) {
        float4 f = p4[i];
        r[4*i+0] = (_Float16)f.x;
        r[4*i+1] = (_Float16)f.y;
        r[4*i+2] = (_Float16)f.z;
        r[4*i+3] = (_Float16)f.w;
    }
    return r;
}

// gfx1250 async DMA: copy 16B global -> LDS without touching VGPR data path.
// GV addressing: %0 = LDS byte offset (VGPR), %1 = 64-bit global addr (VGPR pair).
// Tracked by ASYNCcnt.
__device__ __forceinline__ void async_cp16(unsigned lds_byte_off, const void* g) {
    asm volatile("global_load_async_to_lds_b128 %0, %1, off"
                 :: "v"(lds_byte_off), "v"(g) : "memory");
}
__device__ __forceinline__ void wait_async0() {
    asm volatile("s_wait_asynccnt 0x0" ::: "memory");
}

// ---------------------------------------------------------------------------
// Kernel 1: projection  out = X @ W^T  (X: [8192,256] f32, W: [256,256] f32)
// Result stored f16. vt_mode=0 -> [b,h,n,d] layout (Q,K); 1 -> [b,h,d,n] (V^T)
// One 16x16 output tile per wave, K=256 as 8 chained 16x16x32 f16 WMMAs.
// ---------------------------------------------------------------------------
__global__ __launch_bounds__(256) void proj_kernel(const float* __restrict__ X,
                                                   const float* __restrict__ W,
                                                   _Float16* __restrict__ out,
                                                   float scale, int vt_mode) {
    const int tid  = threadIdx.x;
    const int wave = tid >> 5, lane = tid & 31;
    const int l16  = lane & 15, hs = lane >> 4;          // lane-half select
    const int t    = blockIdx.x * 8 + wave;              // tile id
    const int tm   = t >> 4;                             // 512 row tiles
    const int tn   = t & 15;                             // 16 col tiles
    const int arow = tm * 16 + l16;
    const int bcol = tn * 16 + l16;

    v8f c = {};
#pragma unroll
    for (int kk = 0; kk < 8; ++kk) {
        const int k0 = kk * 32 + hs * 16;
        v16h a  = ld16_f32_to_f16(X + (size_t)arow * DMODEL + k0);
        v16h bm = ld16_f32_to_f16(W + (size_t)bcol * DMODEL + k0);
        c = __builtin_amdgcn_wmma_f32_16x16x32_f16(false, a, false, bm,
                                                   (short)0, c, false, false);
    }

#pragma unroll
    for (int r = 0; r < 8; ++r) {
        const int row = tm * 16 + r + hs * 8;            // C-layout: M = r + 8*half
        const int col = tn * 16 + l16;                   // N = lane%16
        const int b = row >> 11, n = row & 2047;
        const int h = col >> 5,  d = col & 31;
        const _Float16 val = (_Float16)(c[r] * scale);
        size_t idx;
        if (vt_mode) idx = ((size_t)(b * HEADS + h) * DHEAD + d) * NSEQ + n;
        else         idx = ((size_t)(b * HEADS + h) * NSEQ + n) * DHEAD + d;
        out[idx] = val;
    }
}

// ---------------------------------------------------------------------------
// Kernel 2: flash attention with additive positional bias, streamed once.
// Block = 8 wave32; block handles (b, h, 128 query rows). K/V 32-col tiles
// are DMA'd into a double-buffered dynamic-LDS region with
// global_load_async_to_lds_b128 (overlapped with compute), bias is streamed
// with non-temporal loads, scores/PV use f16 WMMA with f32 accumulation.
//
// Dynamic LDS layout (halves):
//   [0    .. 2047]  K tiles,  2 buffers x 1024 (32x32 f16, [j][d])
//   [2048 .. 4095]  V^T tiles, 2 buffers x 1024 (32x32 f16, [d][j])
//   [4096 .. 8191]  per-wave P staging, 8 x 512 (16x32 f16)
// ---------------------------------------------------------------------------
#define ATTN_SMEM_BYTES (8192 * 2)

__device__ __forceinline__ void stage_tiles_async(const _Float16* __restrict__ kh,
                                                  const _Float16* __restrict__ vt,
                                                  size_t bh, int j0, int tid, int sbuf) {
    if (tid < 128) {
        // K tile: contiguous 2 KB, 128 lanes x 16B
        const void* g = kh + (bh * NSEQ + j0) * DHEAD + tid * 8;
        async_cp16((unsigned)(sbuf * 2048 + tid * 16), g);
    } else {
        // V^T tile: 32 rows x 64B, 128 lanes x 16B
        const int u = tid - 128, row = u >> 2, cg = u & 3;
        const void* g = vt + (bh * DHEAD + row) * NSEQ + j0 + cg * 8;
        async_cp16((unsigned)(4096 + sbuf * 2048 + row * 64 + cg * 16), g);
    }
}

__global__ __launch_bounds__(256) void attn_kernel(const _Float16* __restrict__ qh,
                                                   const _Float16* __restrict__ kh,
                                                   const _Float16* __restrict__ vt,
                                                   const float* __restrict__ pb,
                                                   _Float16* __restrict__ oh) {
    extern __shared__ __align__(64) _Float16 smem[];

    const int tid  = threadIdx.x;
    const int wave = tid >> 5, lane = tid & 31;
    const int l16  = lane & 15, hs = lane >> 4;
    const int ib   = blockIdx.x & 15;
    const int h    = (blockIdx.x >> 4) & 7;
    const int b    = blockIdx.x >> 7;
    const int i0   = ib * 128 + wave * 16;
    const size_t bh = (size_t)(b * HEADS + h);

    // Q A-fragment: 16x32 f16, held in registers for all 64 key tiles
    v16h aq = *(const v16h*)(qh + (bh * NSEQ + i0 + l16) * DHEAD + hs * 16);

    v8f acc0 = {}, acc1 = {};
    float mrow[8], lrow[8];
#pragma unroll
    for (int r = 0; r < 8; ++r) { mrow[r] = -3.0e38f; lrow[r] = 0.0f; }

    _Float16* pbuf = smem + 4096 + wave * 512;
    const float* pbrow = pb + ((size_t)h * NSEQ + i0) * NSEQ;

    // prologue: kick off DMA of the first K/V tile pair into buffer 0
    stage_tiles_async(kh, vt, bh, 0, tid, 0);

    for (int js = 0; js < 64; ++js) {
        const int s  = js & 1;
        const int j0 = js * 32;

        wait_async0();          // this wave's DMA into buffer s complete
        __syncthreads();        // all waves' DMA complete

        // overlap: start DMA of the next tile pair into the other buffer
        if (js + 1 < 64) stage_tiles_async(kh, vt, bh, j0 + 32, tid, s ^ 1);

        const _Float16* kbuf = smem + s * 1024;
        const _Float16* vbuf = smem + 2048 + s * 1024;

        // --- scores: sim = Q @ K^T over full head dim, one WMMA per j-half -
        v16h bk0 = *(const v16h*)(kbuf + l16 * 32 + hs * 16);
        v16h bk1 = *(const v16h*)(kbuf + (16 + l16) * 32 + hs * 16);
        v8f z = {};
        v8f s0 = __builtin_amdgcn_wmma_f32_16x16x32_f16(false, aq, false, bk0,
                                                        (short)0, z, false, false);
        v8f s1 = __builtin_amdgcn_wmma_f32_16x16x32_f16(false, aq, false, bk1,
                                                        (short)0, z, false, false);

        // --- additive positional bias: 134 MB stream, read once, NT hint ---
#pragma unroll
        for (int r = 0; r < 8; ++r) {
            const float* prow = pbrow + (size_t)(r + hs * 8) * NSEQ + j0;
            s0[r] += __builtin_nontemporal_load(prow + l16);
            s1[r] += __builtin_nontemporal_load(prow + 16 + l16);
            __builtin_prefetch(prow + 32 + l16, 0, 0);   // next j-tile ahead
        }

        // --- online softmax (rows live in 16-lane halves of the wave) ------
#pragma unroll
        for (int r = 0; r < 8; ++r) {
            float tmax = fmaxf(s0[r], s1[r]);
            tmax = fmaxf(tmax, __shfl_xor(tmax, 1, 16));
            tmax = fmaxf(tmax, __shfl_xor(tmax, 2, 16));
            tmax = fmaxf(tmax, __shfl_xor(tmax, 4, 16));
            tmax = fmaxf(tmax, __shfl_xor(tmax, 8, 16));
            const float mn    = fmaxf(mrow[r], tmax);
            const float alpha = __expf(mrow[r] - mn);
            const float p0 = __expf(s0[r] - mn);
            const float p1 = __expf(s1[r] - mn);
            float rs = p0 + p1;
            rs += __shfl_xor(rs, 1, 16);
            rs += __shfl_xor(rs, 2, 16);
            rs += __shfl_xor(rs, 4, 16);
            rs += __shfl_xor(rs, 8, 16);
            lrow[r] = lrow[r] * alpha + rs;
            mrow[r] = mn;
            acc0[r] *= alpha;
            acc1[r] *= alpha;
            const int m = r + hs * 8;                    // C-layout row
            pbuf[m * 32 + l16]      = (_Float16)p0;
            pbuf[m * 32 + 16 + l16] = (_Float16)p1;
        }

        // --- PV: P (C-layout -> A-layout via LDS) times V^T B-fragments ----
        v16h ap  = *(const v16h*)(pbuf + l16 * 32 + hs * 16);
        v16h bv0 = *(const v16h*)(vbuf + l16 * 32 + hs * 16);
        v16h bv1 = *(const v16h*)(vbuf + (16 + l16) * 32 + hs * 16);
        acc0 = __builtin_amdgcn_wmma_f32_16x16x32_f16(false, ap, false, bv0,
                                                      (short)0, acc0, false, false);
        acc1 = __builtin_amdgcn_wmma_f32_16x16x32_f16(false, ap, false, bv1,
                                                      (short)0, acc1, false, false);
        __syncthreads();        // buffer s fully consumed before reuse
    }

    // --- normalize and emit merged-head layout [b, n, h*32+d] ---------------
#pragma unroll
    for (int r = 0; r < 8; ++r) {
        const float inv = 1.0f / lrow[r];
        const int row = i0 + r + hs * 8;
        const size_t base = (size_t)(b * NSEQ + row) * DMODEL + h * DHEAD;
        oh[base + l16]      = (_Float16)(acc0[r] * inv);
        oh[base + 16 + l16] = (_Float16)(acc1[r] * inv);
    }
}

// ---------------------------------------------------------------------------
// Kernel 3: output projection  out = Oh @ Wo^T, f32 result.
// ---------------------------------------------------------------------------
__global__ __launch_bounds__(256) void oproj_kernel(const _Float16* __restrict__ A,
                                                    const float* __restrict__ W,
                                                    float* __restrict__ out) {
    const int tid  = threadIdx.x;
    const int wave = tid >> 5, lane = tid & 31;
    const int l16  = lane & 15, hs = lane >> 4;
    const int t    = blockIdx.x * 8 + wave;
    const int tm   = t >> 4, tn = t & 15;
    const int arow = tm * 16 + l16;
    const int bcol = tn * 16 + l16;

    v8f c = {};
#pragma unroll
    for (int kk = 0; kk < 8; ++kk) {
        const int k0 = kk * 32 + hs * 16;
        v16h a  = *(const v16h*)(A + (size_t)arow * DMODEL + k0);
        v16h bm = ld16_f32_to_f16(W + (size_t)bcol * DMODEL + k0);
        c = __builtin_amdgcn_wmma_f32_16x16x32_f16(false, a, false, bm,
                                                   (short)0, c, false, false);
    }
#pragma unroll
    for (int r = 0; r < 8; ++r) {
        const int row = tm * 16 + r + hs * 8;
        const int col = tn * 16 + l16;
        out[(size_t)row * DMODEL + col] = c[r];
    }
}

// ---------------------------------------------------------------------------
extern "C" void kernel_launch(void* const* d_in, const int* in_sizes, int n_in,
                              void* d_out, int out_size, void* d_ws, size_t ws_size,
                              hipStream_t stream) {
    const float* q  = (const float*)d_in[0];
    const float* k  = (const float*)d_in[1];
    const float* v  = (const float*)d_in[2];
    const float* pb = (const float*)d_in[3];
    const float* Wq = (const float*)d_in[4];
    const float* Wk = (const float*)d_in[5];
    const float* Wv = (const float*)d_in[6];
    const float* Wo = (const float*)d_in[7];

    const size_t per = (size_t)NB * HEADS * NSEQ * DHEAD;   // 2M f16 = 4 MB each
    _Float16* qh = (_Float16*)d_ws;
    _Float16* kh = qh + per;
    _Float16* vt = kh + per;
    _Float16* oh = vt + per;                                // 16 MB total scratch

    const float scale = 0.17677669529663687f;               // 32^-0.5

    dim3 blk(256);
    proj_kernel<<<1024, blk, 0, stream>>>(q, Wq, qh, scale, 0);
    proj_kernel<<<1024, blk, 0, stream>>>(k, Wk, kh, 1.0f, 0);
    proj_kernel<<<1024, blk, 0, stream>>>(v, Wv, vt, 1.0f, 1);
    attn_kernel<<<512, blk, ATTN_SMEM_BYTES, stream>>>(qh, kh, vt, pb, oh);
    oproj_kernel<<<1024, blk, 0, stream>>>(oh, Wo, (float*)d_out);
}
